// GraphCNNCatUnet2_83545703842054
// MI455X (gfx1250) — matrix operand order repaired and checked by reading, not compiled, not run on hardware
//
#include <hip/hip_runtime.h>
#include <hip/hip_bf16.h>
#include <math.h>

// ---------------------------------------------------------------------------
// GraphCNN Cat-UNet on gfx1250 (MI455X).
//  - Chebyshev conv: L = -A/8 (8-neighbor circulant) => 8-point wrap stencil
//    computed in LDS; fused with WMMA bf16 GEMM (f32 accum).
//  - Weights pre-transposed/padded to bf16 once per call (WT[FoutPad][KWpad]),
//    so per-chunk W staging is pure 16B global->LDS vector copies, no guards.
//  - Per block: 64 pixels (4 row-tiles) x 64 output channels; W chunk staged
//    once in LDS and reused by all 4 row-tiles.
//  - WMMA loop double-buffered (two fragment sets, no register rotation):
//    each WMMA waits only on its own set's ds_loads; the other set's loads
//    stay outstanding underneath it.
// ---------------------------------------------------------------------------

typedef __attribute__((ext_vector_type(16))) __bf16 v16bf;
typedef __attribute__((ext_vector_type(8)))  float  v8f;

#define BATCH   32
#define NTOP    3072
#define CHEB_K  6
#define EXT     56      // 16 + 2*HALO
#define HALO    20      // 4 * (CHEB_K - 1)
#define KPAD    392     // bf16 stride (784B = 196 dwords; 196 % 64 == 4)
#define ROWT    4       // row tiles (of 16 pixels) per block

#define SMEM_BYTES ((16 + 64) * KPAD * 2 + 2 * EXT * 64 * 4)   // 91,392 B

__device__ __forceinline__ int wrapn(int n, int Nl)
{
    while (n < 0)   n += Nl;
    while (n >= Nl) n -= Nl;
    return n;
}

__device__ __forceinline__ void load_a_frag(const __bf16* __restrict__ aRow,
                                            int st, int h, v16bf& af)
{
    const int ab = (st << 5) + (h << 3);   // K = st*32 + 16*(e/8) + 8h + e%8
    #pragma unroll
    for (int e = 0; e < 8; ++e) af[e]     = aRow[ab + e];
    #pragma unroll
    for (int e = 0; e < 8; ++e) af[8 + e] = aRow[ab + 16 + e];
}

__device__ __forceinline__ void load_b_frag(const __bf16* __restrict__ bRow,
                                            int st, int h, v16bf& bf)
{
    const int bb = (st << 5) + (h << 3);   // K = st*32 + 16*(e/8) + 8h + e%8
    #pragma unroll
    for (int e = 0; e < 8; ++e) bf[e]     = bRow[bb + e];
    #pragma unroll
    for (int e = 0; e < 8; ++e) bf[8 + e] = bRow[bb + 16 + e];
}

// ---------------------------------------------------------------------------
// Weight prep: W [KFin, Fout] f32  ->  WT [FoutPad, KWpad] bf16 (transposed,
// zero-padded).  WT[c][kw] = bf16(W[kw][c]).
// ---------------------------------------------------------------------------
__global__ void prep_w_kernel(const float* __restrict__ W, __bf16* __restrict__ WT,
                              int KFin, int Fout, int KWpad, int FoutPad)
{
    int i = blockIdx.x * blockDim.x + threadIdx.x;
    int total = FoutPad * KWpad;
    if (i >= total) return;
    int kw = i % KWpad;
    int c  = i / KWpad;
    float v = (c < Fout && kw < KFin) ? W[(size_t)kw * Fout + c] : 0.f;
    WT[i] = (__bf16)v;
}

// ---------------------------------------------------------------------------
// Xin : [B, Nl, Fin] fp32, Out : [B, Nl, Fout] fp32
// WT  : [FoutPad, KWpad] bf16 (row c = output channel, col kw = k*Fin + f)
// CFL : log2 of per-chunk channel count (2 when Fin==4, else 6)
// act : 1 = relu, 2 = softplus
// ---------------------------------------------------------------------------
__global__ __launch_bounds__(128) void cheb_gemm_kernel(
    const float* __restrict__ Xin, const __bf16* __restrict__ WT,
    const float* __restrict__ bias, float* __restrict__ Out,
    int Nl, int Fin, int Fout, int KWpad, int CFL, int act)
{
    extern __shared__ char smem[];
    __bf16* sAs = (__bf16*)smem;                          // 16  x KPAD bf16
    __bf16* sBs = (__bf16*)(smem + 16 * KPAD * 2);        // 64  x KPAD bf16
    float*  sT0 = (float*)(smem + (16 + 64) * KPAD * 2);  // EXT x 64   f32
    float*  sT1 = sT0 + EXT * 64;                         // EXT x 64   f32

    const int tid  = threadIdx.x;
    const int lane = tid & 31;            // wave32
    const int wave = tid >> 5;            // 4 waves / block
    const int h    = lane >> 4;
    const int l16  = lane & 15;

    const int rowBlocks = (Nl + 63) >> 6;
    const int b    = blockIdx.x / rowBlocks;
    const int n0b  = (blockIdx.x % rowBlocks) << 6;       // 64-pixel block
    const int col0 = blockIdx.y * 64;
    const int colLocal = wave * 16 + l16;
    const int ncol = col0 + colLocal;

    const int CF     = 1 << CFL;          // 4 or 64
    const int CFM    = CF - 1;
    const int nCh    = Fin >> CFL;
    const int KC     = CHEB_K * CF;       // valid staged K per chunk
    const int KSTEPS = (KC + 31) >> 5;
    const int padK   = KSTEPS << 5;

    v8f acc[ROWT];
    #pragma unroll
    for (int rt = 0; rt < ROWT; ++rt)
        acc[rt] = (v8f){0.f, 0.f, 0.f, 0.f, 0.f, 0.f, 0.f, 0.f};

    const __bf16* aRow = sAs + l16 * KPAD;
    const __bf16* bRow = sBs + colLocal * KPAD;

    for (int ch = 0; ch < nCh; ++ch) {
        const int f0 = ch << CFL;

        __syncthreads();   // previous chunk's WMMA reads of sAs/sBs complete

        // ---- stage W chunk -> sBs: pure 16B copies (WT is padded bf16) ----
        if (CF == 64) {
            // 64 cols x 6 runs (of 64 bf16) x 8 uint4; iterate 8 run slots, skip >=6
            for (int idx = tid; idx < 4096; idx += 128) {
                int col = idx >> 6;
                int r   = (idx >> 3) & 7;
                int u   = idx & 7;
                if (r < CHEB_K) {
                    const uint4* src = (const uint4*)(WT + (size_t)(col0 + col) * KWpad
                                                         + r * Fin + f0 + (u << 3));
                    *((uint4*)(sBs + col * KPAD + (r << 6) + (u << 3))) = *src;
                }
            }
        } else {
            // CF == 4 (conv1): padK == 32 => 4 uint4 per col, WT row is 32 bf16
            for (int idx = tid; idx < 256; idx += 128) {
                int col = idx >> 2;
                int u   = idx & 3;
                const uint4* src = (const uint4*)(WT + (size_t)(col0 + col) * KWpad + (u << 3));
                *((uint4*)(sBs + col * KPAD + (u << 3))) = *src;
            }
        }
        __syncthreads();

        #pragma unroll
        for (int rt = 0; rt < ROWT; ++rt) {
            const int n0 = n0b + (rt << 4);
            if (rt) __syncthreads();  // prior rt's WMMA reads of sAs complete

            // ---- load T0 over extended window (wrap mod Nl) ----
            for (int idx = tid; idx < EXT * CF; idx += 128) {
                int i = idx >> CFL, f = idx & CFM;
                int n = wrapn(n0 - HALO + i, Nl);
                sT0[i * CF + f] = Xin[((size_t)b * Nl + n) * Fin + f0 + f];
            }
            // ---- zero-pad staged A cols [KC, padK) (only when CF == 4) ----
            if (padK > KC) {
                for (int idx = tid; idx < 16 * 8; idx += 128) {
                    int m = idx >> 3, kc = KC + (idx & 7);
                    sAs[m * KPAD + kc] = (__bf16)0.0f;
                }
            }
            __syncthreads();

            // ---- stage T0 central; T1 = L*T0 = -0.125 * sum(neighbors) ----
            for (int idx = tid; idx < 16 * CF; idx += 128) {
                int m = idx >> CFL, f = idx & CFM;
                sAs[m * KPAD + f] = (__bf16)sT0[(HALO + m) * CF + f];
            }
            for (int idx = tid; idx < (EXT - 8) * CF; idx += 128) {
                int i = 4 + (idx >> CFL), f = idx & CFM;
                float s = sT0[(i-4)*CF+f] + sT0[(i-3)*CF+f] + sT0[(i-2)*CF+f] + sT0[(i-1)*CF+f]
                        + sT0[(i+1)*CF+f] + sT0[(i+2)*CF+f] + sT0[(i+3)*CF+f] + sT0[(i+4)*CF+f];
                sT1[i * CF + f] = -0.125f * s;
            }
            __syncthreads();

            // ---- Chebyshev recurrence, in-place ping-pong ----
            float* prev = sT0;   // T_{k-2}
            float* cur  = sT1;   // T_{k-1}
            for (int kk = 2; kk < CHEB_K; ++kk) {
                for (int idx = tid; idx < 16 * CF; idx += 128) {
                    int m = idx >> CFL, f = idx & CFM;
                    sAs[m * KPAD + ((kk - 1) << CFL) + f] = (__bf16)cur[(HALO + m) * CF + f];
                }
                int lo  = kk << 2;
                int cnt = (EXT - (kk << 3)) * CF;
                for (int idx = tid; idx < cnt; idx += 128) {
                    int i = lo + (idx >> CFL), f = idx & CFM;
                    float s = cur[(i-4)*CF+f] + cur[(i-3)*CF+f] + cur[(i-2)*CF+f] + cur[(i-1)*CF+f]
                            + cur[(i+1)*CF+f] + cur[(i+2)*CF+f] + cur[(i+3)*CF+f] + cur[(i+4)*CF+f];
                    prev[i * CF + f] = -0.25f * s - prev[i * CF + f];  // elementwise, race-free
                }
                __syncthreads();
                float* t = prev; prev = cur; cur = t;
            }
            for (int idx = tid; idx < 16 * CF; idx += 128) {   // stage T5 central
                int m = idx >> CFL, f = idx & CFM;
                sAs[m * KPAD + ((CHEB_K - 1) << CFL) + f] = (__bf16)cur[(HALO + m) * CF + f];
            }
            __syncthreads();

            // ---- double-buffered WMMA: acc[rt] += A(16 x padK) * B(padK x 16) ----
            {
                v16bf a0, b0, a1, b1;
                load_a_frag(aRow, 0, h, a0);
                load_b_frag(bRow, 0, h, b0);
                int st = 0;
                for (; st + 2 < KSTEPS; st += 2) {
                    load_a_frag(aRow, st + 1, h, a1);
                    load_b_frag(bRow, st + 1, h, b1);
                    acc[rt] = __builtin_amdgcn_wmma_f32_16x16x32_bf16(
                        false, a0, false, b0, (short)0, acc[rt], false, false);
                    load_a_frag(aRow, st + 2, h, a0);
                    load_b_frag(bRow, st + 2, h, b0);
                    acc[rt] = __builtin_amdgcn_wmma_f32_16x16x32_bf16(
                        false, a1, false, b1, (short)0, acc[rt], false, false);
                }
                if (st + 1 < KSTEPS) {   // two steps remain
                    load_a_frag(aRow, st + 1, h, a1);
                    load_b_frag(bRow, st + 1, h, b1);
                    acc[rt] = __builtin_amdgcn_wmma_f32_16x16x32_bf16(
                        false, a0, false, b0, (short)0, acc[rt], false, false);
                    acc[rt] = __builtin_amdgcn_wmma_f32_16x16x32_bf16(
                        false, a1, false, b1, (short)0, acc[rt], false, false);
                } else {                  // one step remains (conv1)
                    acc[rt] = __builtin_amdgcn_wmma_f32_16x16x32_bf16(
                        false, a0, false, b0, (short)0, acc[rt], false, false);
                }
            }
        }
    }

    // ---- store C tiles: lane col = lane%16, row M = v + 8*h ----
    if (ncol < Fout) {
        float bv = bias[ncol];
        #pragma unroll
        for (int rt = 0; rt < ROWT; ++rt) {
            #pragma unroll
            for (int v = 0; v < 8; ++v) {
                int row = n0b + (rt << 4) + v + (h << 3);
                if (row < Nl) {
                    float val = acc[rt][v] + bv;
                    if (act == 1)      val = fmaxf(val, 0.f);
                    else if (act == 2) val = (val > 20.f) ? val : log1pf(expf(val));
                    Out[((size_t)b * Nl + row) * Fout + ncol] = val;
                }
            }
        }
    }
}

// ---------------------------------------------------------------------------
// x [B, S, N4] -> xt [B, N4, S]
// ---------------------------------------------------------------------------
__global__ void transpose_in_kernel(const float* __restrict__ x, float* __restrict__ xt)
{
    int i = blockIdx.x * blockDim.x + threadIdx.x;
    int total = BATCH * NTOP * 4;
    if (i >= total) return;
    int s = i & 3;
    int n = (i >> 2) % NTOP;
    int b = i / (NTOP * 4);
    xt[i] = x[((size_t)b * 4 + s) * NTOP + n];
}

// ---------------------------------------------------------------------------
// max-pool by 4 over pixel dim.  In [B, 4*Ng, F] -> Ov [B, Ng, F], Oi (argmax)
// ---------------------------------------------------------------------------
__global__ void pool4_kernel(const float* __restrict__ In, float* __restrict__ Ov,
                             unsigned char* __restrict__ Oi, int Ng, int F)
{
    int i = blockIdx.x * blockDim.x + threadIdx.x;
    int total = BATCH * Ng * F;
    if (i >= total) return;
    int f = i % F;
    int g = (i / F) % Ng;
    int b = i / (F * Ng);
    const float* base = In + ((size_t)b * (4 * Ng) + 4 * g) * F + f;
    float best = base[0]; int bi = 0;
    #pragma unroll
    for (int j = 1; j < 4; ++j) {
        float v = base[(size_t)j * F];
        if (v > best) { best = v; bi = j; }
    }
    Ov[i] = best;
    Oi[i] = (unsigned char)bi;
}

// ---------------------------------------------------------------------------
// Out[b,n,:] = concat( unpool(Pv,Idx)[b,n, 0..Fup), Sk[b,n, 0..Fsk) )
// ---------------------------------------------------------------------------
__global__ void unpool_concat_kernel(const float* __restrict__ Pv,
                                     const unsigned char* __restrict__ Idx,
                                     const float* __restrict__ Sk,
                                     float* __restrict__ Out,
                                     int N, int Fup, int Fsk)
{
    int Fc = Fup + Fsk;
    int i = blockIdx.x * blockDim.x + threadIdx.x;
    int total = BATCH * N * Fc;
    if (i >= total) return;
    int c = i % Fc;
    int n = (i / Fc) % N;
    int b = i / (Fc * N);
    float v;
    if (c < Fup) {
        int g = n >> 2, j = n & 3;
        size_t pi = ((size_t)b * (N >> 2) + g) * Fup + c;
        v = (Idx[pi] == (unsigned char)j) ? Pv[pi] : 0.f;
    } else {
        v = Sk[((size_t)b * N + n) * Fsk + (c - Fup)];
    }
    Out[i] = v;
}

// ---------------------------------------------------------------------------
// fodf [B, N4, 3] (softplus applied) -> out = [ fodf[:,:,0] | max_n ch1 | max_n ch2 ]
// ---------------------------------------------------------------------------
__global__ void final_kernel(const float* __restrict__ F, float* __restrict__ out)
{
    __shared__ float s2[256];
    __shared__ float s3[256];
    int b = blockIdx.x;
    float m2 = -3.4e38f, m3 = -3.4e38f;
    for (int n = threadIdx.x; n < NTOP; n += blockDim.x) {
        const float* p = F + ((size_t)b * NTOP + n) * 3;
        out[(size_t)b * NTOP + n] = p[0];
        m2 = fmaxf(m2, p[1]);
        m3 = fmaxf(m3, p[2]);
    }
    s2[threadIdx.x] = m2; s3[threadIdx.x] = m3;
    __syncthreads();
    for (int st = 128; st > 0; st >>= 1) {
        if (threadIdx.x < st) {
            s2[threadIdx.x] = fmaxf(s2[threadIdx.x], s2[threadIdx.x + st]);
            s3[threadIdx.x] = fmaxf(s3[threadIdx.x], s3[threadIdx.x + st]);
        }
        __syncthreads();
    }
    if (threadIdx.x == 0) {
        out[BATCH * NTOP + b]         = s2[0];
        out[BATCH * NTOP + BATCH + b] = s3[0];
    }
}

// ---------------------------------------------------------------------------
// Host-side orchestration
// ---------------------------------------------------------------------------
extern "C" void kernel_launch(void* const* d_in, const int* in_sizes, int n_in,
                              void* d_out, int out_size, void* d_ws, size_t ws_size,
                              hipStream_t stream)
{
    (void)in_sizes; (void)n_in; (void)out_size; (void)ws_size;

    // allow >64KB dynamic LDS (CDNA5: up to 320KB per workgroup)
    (void)hipFuncSetAttribute((const void*)cheb_gemm_kernel,
                              hipFuncAttributeMaxDynamicSharedMemorySize, SMEM_BYTES);

    const float* x = (const float*)d_in[0];
    // d_in: [0]=x, [1..5]=laps (unused: L is a known circulant), [6..]=W,b pairs
    const float* Wp[19]; const float* bp[19];
    for (int i = 0; i < 19; ++i) {
        Wp[i] = (const float*)d_in[6 + 2 * i];
        bp[i] = (const float*)d_in[7 + 2 * i];
    }

    // layer shapes: {Fin, Fout}
    static const int LFin[19]  = {   4,  64,  64, 128, 128, 256, 256, 512,  512, 1024,
                                  1024, 512, 512, 256, 256, 128, 128,  64,   64 };
    static const int LFout[19] = {  64,  64, 128, 128, 256, 256, 512, 512, 1024,  512,
                                   512, 256, 256, 128, 128,  64,  64,  64,    3 };

    char* ws = (char*)d_ws;
    size_t off = 0;
    auto alloc = [&](size_t bytes) -> char* {
        char* p = ws + off;
        off += (bytes + 255) & ~(size_t)255;
        return p;
    };

    float* XT = (float*)alloc((size_t)BATCH * NTOP * 4   * sizeof(float));  // [B,3072,4]
    float* A  = (float*)alloc((size_t)BATCH * NTOP * 128 * sizeof(float));  // big ping
    float* Bb = (float*)alloc((size_t)BATCH * NTOP * 64  * sizeof(float));  // big pong
    float* E4 = (float*)alloc((size_t)BATCH * NTOP * 64  * sizeof(float));  // skip lvl4
    float* E3 = (float*)alloc((size_t)BATCH * 768  * 128 * sizeof(float));  // skip lvl3
    float* E2 = (float*)alloc((size_t)BATCH * 192  * 256 * sizeof(float));  // skip lvl2
    float* E1 = (float*)alloc((size_t)BATCH * 48   * 512 * sizeof(float));  // skip lvl1
    unsigned char* I3 = (unsigned char*)alloc((size_t)BATCH * 768 * 64);
    unsigned char* I2 = (unsigned char*)alloc((size_t)BATCH * 192 * 128);
    unsigned char* I1 = (unsigned char*)alloc((size_t)BATCH * 48  * 256);
    unsigned char* IB = (unsigned char*)alloc((size_t)BATCH * 12  * 512);

    // ---- weight prep: transpose + pad + convert to bf16 ----
    __bf16* WT[19];
    int KWp[19];
    for (int i = 0; i < 19; ++i) {
        int KFin    = CHEB_K * LFin[i];
        int KWpad   = ((KFin + 31) / 32) * 32;
        int FoutPad = ((LFout[i] + 63) / 64) * 64;
        KWp[i] = KWpad;
        WT[i]  = (__bf16*)alloc((size_t)FoutPad * KWpad * sizeof(__bf16));
        int total = FoutPad * KWpad;
        prep_w_kernel<<<(total + 255) / 256, 256, 0, stream>>>(
            Wp[i], WT[i], KFin, LFout[i], KWpad, FoutPad);
    }

    auto cheb = [&](const float* in, int li, float* out, int Nl, int act) {
        int Fin = LFin[li], Fout = LFout[li];
        dim3 grid(BATCH * ((Nl + 63) / 64), (Fout + 63) / 64);
        int cfl = (Fin < 64) ? 2 : 6;
        cheb_gemm_kernel<<<grid, 128, SMEM_BYTES, stream>>>(
            in, WT[li], bp[li], out, Nl, Fin, Fout, KWp[li], cfl, act);
    };
    auto pool = [&](const float* in, float* ov, unsigned char* oi, int Ng, int F) {
        int total = BATCH * Ng * F;
        pool4_kernel<<<(total + 255) / 256, 256, 0, stream>>>(in, ov, oi, Ng, F);
    };
    auto upcat = [&](const float* pv, const unsigned char* ip, const float* sk,
                     float* out, int N, int Fup, int Fsk) {
        int total = BATCH * N * (Fup + Fsk);
        unpool_concat_kernel<<<(total + 255) / 256, 256, 0, stream>>>(pv, ip, sk, out, N, Fup, Fsk);
    };

    // encoder
    transpose_in_kernel<<<(BATCH * NTOP * 4 + 255) / 256, 256, 0, stream>>>(x, XT);
    cheb(XT, 0, A,  3072, 1);   // conv1 (store relu(x1))
    cheb(A,  1, E4, 3072, 1);   // enc4
    pool(E4, A, I3, 768, 64);
    cheb(A,  2, Bb,  768, 1);   // enc3
    cheb(Bb, 3, E3,  768, 1);   // enc3_2
    pool(E3, A, I2, 192, 128);
    cheb(A,  4, Bb,  192, 1);   // enc2
    cheb(Bb, 5, E2,  192, 1);   // enc2_2
    pool(E2, A, I1, 48, 256);
    cheb(A,  6, Bb,   48, 1);   // enc1
    cheb(Bb, 7, E1,   48, 1);   // enc1_2
    pool(E1, A, IB, 12, 512);
    cheb(A,  8, Bb,   12, 1);   // bot
    cheb(Bb, 9, A,    12, 1);   // bot2

    // decoder
    upcat(A,  IB, E1, Bb,   48, 512, 512);  // d1cat
    cheb(Bb, 10, A,    48, 1);  // dec1
    cheb(A,  11, Bb,   48, 1);  // dec1_2
    upcat(Bb, I1, E2, A,   192, 256, 256);  // d2cat
    cheb(A,  12, Bb,  192, 1);  // dec2
    cheb(Bb, 13, A,   192, 1);  // dec2_2
    upcat(A,  I2, E3, Bb,  768, 128, 128);  // d3cat
    cheb(Bb, 14, A,   768, 1);  // dec3
    cheb(A,  15, Bb,  768, 1);  // dec3_2
    upcat(Bb, I3, E4, A,  3072,  64,  64);  // d4cat
    cheb(A,  16, Bb, 3072, 1);  // dec4
    cheb(Bb, 17, A,  3072, 1);  // dec4_2
    cheb(A,  18, Bb, 3072, 2);  // pred (softplus)

    final_kernel<<<BATCH, 256, 0, stream>>>(Bb, (float*)d_out);
}
